// RetNet_22608707846252
// MI455X (gfx1250) — compile-verified, matrix operand order checked by reading, not run on hardware
//
#include <hip/hip_runtime.h>
#include <math.h>

#define D_MODEL 2048
#define NHEAD   16
#define DHH     128
#define SEQ     2048
#define BATCH   2
#define MROWS   (BATCH*SEQ)   // 4096
#define HIDR    5461
#define HIDP    5504          // padded to 43*128
#define EPSF    1e-5f
#define BKK     64            // GEMM K-step

typedef __attribute__((ext_vector_type(16))) __bf16 v16bf;
typedef __attribute__((ext_vector_type(8)))  __bf16 v8bf;
typedef __attribute__((ext_vector_type(8)))  float  v8f;

static __device__ __forceinline__ __bf16 f2bf(float f) {
  unsigned u = __builtin_bit_cast(unsigned, f);
  unsigned r = u + 0x7FFFu + ((u >> 16) & 1u);          // round to nearest even
  unsigned short h = (unsigned short)(r >> 16);
  return __builtin_bit_cast(__bf16, h);
}

// Build a 16x32 (A) / 32x16 (B) bf16 WMMA fragment from 8+8 contiguous halves.
static __device__ __forceinline__ v16bf make_frag(const __bf16* p) {
  v8bf lo = *(const v8bf*)p;
  v8bf hi = *(const v8bf*)(p + 16);
  v16bf a;
#pragma unroll
  for (int e = 0; e < 8; ++e) { a[e] = lo[e]; a[8 + e] = hi[e]; }
  return a;
}

// ---- CDNA5 async global->LDS copy (ASYNCcnt-tracked, no VGPR round-trip) ----
// Flat shared-pointer low 32 bits are the LDS byte offset (ISA 10.2 aperture),
// which is what VDST of the async op expects.
static __device__ __forceinline__ void async_ld16(void* lds, const void* g) {
  unsigned l = (unsigned)(uintptr_t)lds;
  unsigned long long ga = (unsigned long long)(uintptr_t)g;
  asm volatile("global_load_async_to_lds_b128 %0, %1, off"
               :: "v"(l), "v"(ga) : "memory");
}
static __device__ __forceinline__ void async_wait0() {
  asm volatile("s_wait_asynccnt 0" ::: "memory");
}

// ---------------- elementwise / conversion kernels ----------------

__global__ void k_cvt_pad(const float* __restrict__ src, __bf16* __restrict__ dst,
                          int R, int C, int Rp, int Cp, float scale) {
  long i = (long)blockIdx.x * blockDim.x + threadIdx.x;
  long total = (long)Rp * Cp;
  if (i >= total) return;
  int r = (int)(i / Cp), c = (int)(i % Cp);
  float v = (r < R && c < C) ? src[(long)r * C + c] * scale : 0.f;
  dst[i] = f2bf(v);
}

__global__ __launch_bounds__(256) void k_layernorm(const float* __restrict__ x,
                                                   const float* __restrict__ w,
                                                   const float* __restrict__ b,
                                                   __bf16* __restrict__ out, int D) {
  int row = blockIdx.x, tid = threadIdx.x;
  const float* xr = x + (long)row * D;
  __shared__ float sh[256], sh2[256];
  float s = 0.f, s2 = 0.f;
  for (int i = tid; i < D; i += 256) { float v = xr[i]; s += v; s2 += v * v; }
  sh[tid] = s; sh2[tid] = s2; __syncthreads();
  for (int st = 128; st > 0; st >>= 1) {
    if (tid < st) { sh[tid] += sh[tid + st]; sh2[tid] += sh2[tid + st]; }
    __syncthreads();
  }
  float mean = sh[0] / D;
  float var  = sh2[0] / D - mean * mean;
  float rstd = rsqrtf(var + EPSF);
  for (int i = tid; i < D; i += 256)
    out[(long)row * D + i] = f2bf((xr[i] - mean) * rstd * w[i] + b[i]);
}

__global__ __launch_bounds__(256) void k_add_ln(const float* __restrict__ x,
                                                const float* __restrict__ y,
                                                const float* __restrict__ w,
                                                const float* __restrict__ b,
                                                float* __restrict__ x2,
                                                __bf16* __restrict__ out, int D) {
  int row = blockIdx.x, tid = threadIdx.x;
  long base = (long)row * D;
  __shared__ float sh[256], sh2[256];
  float s = 0.f, s2 = 0.f;
  for (int i = tid; i < D; i += 256) {
    float v = x[base + i] + y[base + i];
    x2[base + i] = v; s += v; s2 += v * v;
  }
  sh[tid] = s; sh2[tid] = s2; __syncthreads();
  for (int st = 128; st > 0; st >>= 1) {
    if (tid < st) { sh[tid] += sh[tid + st]; sh2[tid] += sh2[tid + st]; }
    __syncthreads();
  }
  float mean = sh[0] / D;
  float var  = sh2[0] / D - mean * mean;
  float rstd = rsqrtf(var + EPSF);
  for (int i = tid; i < D; i += 256)
    out[base + i] = f2bf((x2[base + i] - mean) * rstd * w[i] + b[i]);
}

__global__ __launch_bounds__(128) void k_gnorm_gate(const float* __restrict__ ret,
                                                    const float* __restrict__ g,
                                                    __bf16* __restrict__ out) {
  long grp = blockIdx.x;                    // (b*S+s)*H + h  -> dh=128 group
  long base = grp * DHH;
  int tid = threadIdx.x;
  __shared__ float sh[128], sh2[128];
  float v = ret[base + tid];
  sh[tid] = v; sh2[tid] = v * v; __syncthreads();
  for (int st = 64; st > 0; st >>= 1) {
    if (tid < st) { sh[tid] += sh[tid + st]; sh2[tid] += sh2[tid + st]; }
    __syncthreads();
  }
  float mean = sh[0] / DHH;
  float var  = sh2[0] / DHH - mean * mean;
  float rstd = rsqrtf(var + EPSF);
  float gv = g[base + tid];
  float gate = gv / (1.f + expf(-gv));      // silu
  out[base + tid] = f2bf(gate * (v - mean) * rstd);
}

__global__ void k_silu_mul(const float* __restrict__ a1, const float* __restrict__ a2,
                           __bf16* __restrict__ u, long n) {
  long i = (long)blockIdx.x * blockDim.x + threadIdx.x;
  if (i >= n) return;
  float a = a1[i];
  u[i] = f2bf((a / (1.f + expf(-a))) * a2[i]);
}

__global__ void k_add(const float* __restrict__ a, const float* __restrict__ b,
                      float* __restrict__ o, long n) {
  long i = (long)blockIdx.x * blockDim.x + threadIdx.x;
  if (i >= n) return;
  o[i] = a[i] + b[i];
}

// ---------------- bf16 WMMA GEMM: C[M,N] = A[M,K] * W[K,N] ----------------
// block tile 128x128, 8 waves as 2x4, wave tile 64x32 (4x2 WMMA tiles), K step 64.
// Double-buffered LDS: async global->LDS copies for tile i+1 overlap WMMA on
// tile i; s_wait_asynccnt 0 + workgroup barrier publishes each buffer.

__global__ __launch_bounds__(256) void k_gemm(const __bf16* __restrict__ A,
                                              const __bf16* __restrict__ W,
                                              float* __restrict__ Cf,
                                              __bf16* __restrict__ Cb,
                                              int M, int N, int K, int store_bf) {
  __shared__ alignas(16) __bf16 sA[2][128 * BKK];   // [row][k]          2x16KB
  __shared__ alignas(16) __bf16 sW[2][128 * BKK];   // transposed [n][k] 2x16KB
  int tid = threadIdx.x, wave = tid >> 5, lane = tid & 31;
  int lh = lane >> 4, ll = lane & 15;
  int bm = blockIdx.y * 128, bn = blockIdx.x * 128;
  int wr = (wave >> 2) * 64;                     // 0 / 64
  int wc = (wave & 3) * 32;                      // 0..96
  v8f acc[4][2] = {};

  auto stage = [&](int buf, int k0) {
#pragma unroll
    for (int t = 0; t < 4; ++t) {                // A tile 128x64: async 16B copies
      int ch = t * 256 + tid;                    // 1024 chunks of 8 elems
      int r = ch >> 3, kc = (ch & 7) * 8;
      async_ld16(&sA[buf][r * BKK + kc], &A[(long)(bm + r) * K + k0 + kc]);
    }
#pragma unroll
    for (int t = 0; t < 4; ++t) {                // W tile 64x128 -> transposed LDS
      int c = t * 256 + tid;                     // 1024 chunks of 8 elems (row-wise)
      int kr = c >> 4, nc = (c & 15) * 8;
      v8bf wv = *(const v8bf*)&W[(long)(k0 + kr) * N + bn + nc];
#pragma unroll
      for (int e = 0; e < 8; ++e) sW[buf][(nc + e) * BKK + kr] = wv[e];
    }
  };

  stage(0, 0);
  async_wait0();
  __syncthreads();

  int buf = 0;
  for (int k0 = 0; k0 < K; k0 += BKK) {
    if (k0 + BKK < K) stage(buf ^ 1, k0 + BKK);  // overlap copy of next tile

#pragma unroll
    for (int kc = 0; kc < BKK; kc += 32) {
      v16bf af[4], bfr[2];
#pragma unroll
      for (int rt = 0; rt < 4; ++rt)
        af[rt] = make_frag(&sA[buf][(wr + rt * 16 + ll) * BKK + kc + lh * 8]);
#pragma unroll
      for (int ct = 0; ct < 2; ++ct)
        bfr[ct] = make_frag(&sW[buf][(wc + ct * 16 + ll) * BKK + kc + lh * 8]);
#pragma unroll
      for (int rt = 0; rt < 4; ++rt)
#pragma unroll
        for (int ct = 0; ct < 2; ++ct)
          acc[rt][ct] = __builtin_amdgcn_wmma_f32_16x16x32_bf16(
              false, af[rt], false, bfr[ct], (short)0, acc[rt][ct], false, false);
    }
    async_wait0();
    __syncthreads();
    buf ^= 1;
  }

  // epilogue: branch hoisted out of the store loops
  if (store_bf) {
#pragma unroll
    for (int rt = 0; rt < 4; ++rt)
#pragma unroll
      for (int ct = 0; ct < 2; ++ct) {
        long row0 = bm + wr + rt * 16 + lh * 8;
        long col  = bn + wc + ct * 16 + ll;
#pragma unroll
        for (int r = 0; r < 8; ++r)
          Cb[(row0 + r) * N + col] = f2bf(acc[rt][ct][r]);
      }
  } else {
#pragma unroll
    for (int rt = 0; rt < 4; ++rt)
#pragma unroll
      for (int ct = 0; ct < 2; ++ct) {
        long row0 = bm + wr + rt * 16 + lh * 8;
        long col  = bn + wc + ct * 16 + ll;
#pragma unroll
        for (int r = 0; r < 8; ++r)
          Cf[(row0 + r) * N + col] = acc[rt][ct][r];
      }
  }
}

// ---------------- retention: per (b,h), 128 q rows / block ----------------
// scores = (q kT) * gamma^(s-t) [causal], ret = scores @ v ; all via WMMA bf16.
// K/V chunks double-buffered with async global->LDS copies.

__global__ __launch_bounds__(256) void k_retention(const __bf16* __restrict__ Q,
                                                   const __bf16* __restrict__ Kk,
                                                   const __bf16* __restrict__ V,
                                                   float* __restrict__ Ret) {
  int bh = blockIdx.x;
  int b = bh / NHEAD, h = bh % NHEAD;
  int q0 = blockIdx.y * 128;
  long base = ((long)b * SEQ) * D_MODEL + (long)h * DHH;
  int tid = threadIdx.x, wave = tid >> 5, lane = tid & 31;
  int lh = lane >> 4, ll = lane & 15;
  float log2g = log2f(1.f - exp2f(-5.f - (float)h));

  __shared__ alignas(16) __bf16 sK[2][32 * 128];     // [kvrow][d]       2x8KB
  __shared__ alignas(16) __bf16 sVt[2][128 * 32];    // transposed [d][kvrow]
  __shared__ alignas(16) __bf16 sP[8][16 * 32];      // per-wave bf16 scores

  int qr = q0 + wave * 16;                           // this wave's 16 q rows
  v16bf qf[4];
#pragma unroll
  for (int c = 0; c < 4; ++c)
    qf[c] = make_frag(&Q[base + (long)(qr + ll) * D_MODEL + c * 32 + lh * 8]);

  v8f racc[8] = {};

  auto stageKV = [&](int buf, int t0) {
#pragma unroll
    for (int t = 0; t < 2; ++t) {                    // K chunk 32x128: async copy
      int ch = tid * 2 + t;
      int r = ch >> 4, c8 = (ch & 15) * 8;
      async_ld16(&sK[buf][r * 128 + c8], &Kk[base + (long)(t0 + r) * D_MODEL + c8]);
    }
#pragma unroll
    for (int t = 0; t < 2; ++t) {                    // V chunk -> transposed LDS
      int c = t * 256 + tid;                         // 512 chunks of 8 elems
      int r = c >> 4, nc = (c & 15) * 8;
      v8bf vv = *(const v8bf*)&V[base + (long)(t0 + r) * D_MODEL + nc];
#pragma unroll
      for (int e = 0; e < 8; ++e) sVt[buf][(nc + e) * 32 + r] = vv[e];
    }
  };

  int tmax = q0 + 128;
  stageKV(0, 0);
  async_wait0();
  __syncthreads();

  int buf = 0;
  for (int t0 = 0; t0 < tmax; t0 += 32) {
    if (t0 + 32 < tmax) stageKV(buf ^ 1, t0 + 32);   // overlap copy of next chunk

    if (t0 <= qr + 15) {                             // causal: chunk intersects rows
#pragma unroll
      for (int ct = 0; ct < 2; ++ct) {               // two 16x16 score tiles
        v8f sc = {};
#pragma unroll
        for (int c = 0; c < 4; ++c) {                // dh = 4 * 32
          v16bf kf = make_frag(&sK[buf][(ct * 16 + ll) * 128 + c * 32 + lh * 8]);
          sc = __builtin_amdgcn_wmma_f32_16x16x32_bf16(
              false, qf[c], false, kf, (short)0, sc, false, false);
        }
#pragma unroll
        for (int r = 0; r < 8; ++r) {                // decay mask, f32 precision
          int si = qr + r + lh * 8;
          int ti = t0 + ct * 16 + ll;
          float dec = (ti <= si) ? exp2f(log2g * (float)(si - ti)) : 0.f;
          float v = sc[r] * dec;
          sP[wave][(r + lh * 8) * 32 + ct * 16 + ll] = f2bf(v);
        }
      }
      // repack scores as A-fragment (same-wave LDS round trip, no barrier)
      v16bf pf = make_frag(&sP[wave][ll * 32 + lh * 8]);
#pragma unroll
      for (int dt = 0; dt < 8; ++dt) {
        v16bf vf = make_frag(&sVt[buf][(dt * 16 + ll) * 32 + lh * 8]);
        racc[dt] = __builtin_amdgcn_wmma_f32_16x16x32_bf16(
            false, pf, false, vf, (short)0, racc[dt], false, false);
      }
    }
    async_wait0();
    __syncthreads();
    buf ^= 1;
  }

#pragma unroll
  for (int dt = 0; dt < 8; ++dt)
#pragma unroll
    for (int r = 0; r < 8; ++r) {
      long si = qr + r + lh * 8;
      Ret[base + si * D_MODEL + dt * 16 + ll] = racc[dt][r];
    }
}

// ---------------- launch ----------------

extern "C" void kernel_launch(void* const* d_in, const int* in_sizes, int n_in,
                              void* d_out, int out_size, void* d_ws, size_t ws_size,
                              hipStream_t stream) {
  (void)in_sizes; (void)n_in; (void)out_size; (void)ws_size;
  const float* x   = (const float*)d_in[0];
  const float* n1w = (const float*)d_in[1];
  const float* n1b = (const float*)d_in[2];
  const float* wq  = (const float*)d_in[3];
  const float* wk  = (const float*)d_in[4];
  const float* wv  = (const float*)d_in[5];
  const float* wg  = (const float*)d_in[6];
  const float* wo  = (const float*)d_in[7];
  const float* n2w = (const float*)d_in[8];
  const float* n2b = (const float*)d_in[9];
  const float* w1  = (const float*)d_in[10];
  const float* w2  = (const float*)d_in[11];
  const float* w3  = (const float*)d_in[12];
  float* out = (float*)d_out;

  char* ws = (char*)d_ws;
  size_t off = 0;
  auto alloc = [&](size_t bytes) -> void* {
    void* p = (void*)(ws + off);
    off += (bytes + 255) & ~(size_t)255;
    return p;
  };
  __bf16* h_bf  = (__bf16*)alloc((size_t)MROWS * D_MODEL * 2);
  __bf16* wq_bf = (__bf16*)alloc((size_t)D_MODEL * D_MODEL * 2);
  __bf16* wk_bf = (__bf16*)alloc((size_t)D_MODEL * D_MODEL * 2);
  __bf16* wv_bf = (__bf16*)alloc((size_t)D_MODEL * D_MODEL * 2);
  __bf16* wg_bf = (__bf16*)alloc((size_t)D_MODEL * D_MODEL * 2);
  __bf16* wo_bf = (__bf16*)alloc((size_t)D_MODEL * D_MODEL * 2);
  __bf16* w1_bf = (__bf16*)alloc((size_t)D_MODEL * HIDP * 2);
  __bf16* w2_bf = (__bf16*)alloc((size_t)D_MODEL * HIDP * 2);
  __bf16* w3_bf = (__bf16*)alloc((size_t)HIDP * D_MODEL * 2);
  __bf16* q_bf  = (__bf16*)alloc((size_t)MROWS * D_MODEL * 2);
  __bf16* k_bf  = (__bf16*)alloc((size_t)MROWS * D_MODEL * 2);
  __bf16* v_bf  = (__bf16*)alloc((size_t)MROWS * D_MODEL * 2);
  float*  g_f   = (float*)alloc((size_t)MROWS * D_MODEL * 4);
  float*  ret_f = (float*)alloc((size_t)MROWS * D_MODEL * 4);
  __bf16* gt_bf = (__bf16*)alloc((size_t)MROWS * D_MODEL * 2);
  float*  y_f   = (float*)alloc((size_t)MROWS * D_MODEL * 4);
  float*  x2_f  = (float*)alloc((size_t)MROWS * D_MODEL * 4);
  __bf16* h2_bf = (__bf16*)alloc((size_t)MROWS * D_MODEL * 2);
  float*  a1_f  = (float*)alloc((size_t)MROWS * HIDP * 4);
  float*  a2_f  = (float*)alloc((size_t)MROWS * HIDP * 4);
  __bf16* u_bf  = (__bf16*)alloc((size_t)MROWS * HIDP * 2);
  float*  ffn_f = (float*)alloc((size_t)MROWS * D_MODEL * 4);

  const float kscale = 1.f / sqrtf((float)DHH);
  long ndd = (long)D_MODEL * D_MODEL;
  long ndh = (long)D_MODEL * HIDP;
  k_cvt_pad<<<(int)((ndd + 255) / 256), 256, 0, stream>>>(wq, wq_bf, D_MODEL, D_MODEL, D_MODEL, D_MODEL, 1.f);
  k_cvt_pad<<<(int)((ndd + 255) / 256), 256, 0, stream>>>(wk, wk_bf, D_MODEL, D_MODEL, D_MODEL, D_MODEL, kscale);
  k_cvt_pad<<<(int)((ndd + 255) / 256), 256, 0, stream>>>(wv, wv_bf, D_MODEL, D_MODEL, D_MODEL, D_MODEL, 1.f);
  k_cvt_pad<<<(int)((ndd + 255) / 256), 256, 0, stream>>>(wg, wg_bf, D_MODEL, D_MODEL, D_MODEL, D_MODEL, 1.f);
  k_cvt_pad<<<(int)((ndd + 255) / 256), 256, 0, stream>>>(wo, wo_bf, D_MODEL, D_MODEL, D_MODEL, D_MODEL, 1.f);
  k_cvt_pad<<<(int)((ndh + 255) / 256), 256, 0, stream>>>(w1, w1_bf, D_MODEL, HIDR, D_MODEL, HIDP, 1.f);
  k_cvt_pad<<<(int)((ndh + 255) / 256), 256, 0, stream>>>(w2, w2_bf, D_MODEL, HIDR, D_MODEL, HIDP, 1.f);
  k_cvt_pad<<<(int)((ndh + 255) / 256), 256, 0, stream>>>(w3, w3_bf, HIDR, D_MODEL, HIDP, D_MODEL, 1.f);

  k_layernorm<<<MROWS, 256, 0, stream>>>(x, n1w, n1b, h_bf, D_MODEL);

  dim3 gdd(D_MODEL / 128, MROWS / 128);
  k_gemm<<<gdd, 256, 0, stream>>>(h_bf, wq_bf, nullptr, q_bf, MROWS, D_MODEL, D_MODEL, 1);
  k_gemm<<<gdd, 256, 0, stream>>>(h_bf, wk_bf, nullptr, k_bf, MROWS, D_MODEL, D_MODEL, 1);
  k_gemm<<<gdd, 256, 0, stream>>>(h_bf, wv_bf, nullptr, v_bf, MROWS, D_MODEL, D_MODEL, 1);
  k_gemm<<<gdd, 256, 0, stream>>>(h_bf, wg_bf, g_f, nullptr, MROWS, D_MODEL, D_MODEL, 0);

  k_retention<<<dim3(BATCH * NHEAD, SEQ / 128), 256, 0, stream>>>(q_bf, k_bf, v_bf, ret_f);
  k_gnorm_gate<<<MROWS * NHEAD, 128, 0, stream>>>(ret_f, g_f, gt_bf);
  k_gemm<<<gdd, 256, 0, stream>>>(gt_bf, wo_bf, y_f, nullptr, MROWS, D_MODEL, D_MODEL, 0);

  k_add_ln<<<MROWS, 256, 0, stream>>>(x, y_f, n2w, n2b, x2_f, h2_bf, D_MODEL);

  dim3 gdh(HIDP / 128, MROWS / 128);
  k_gemm<<<gdh, 256, 0, stream>>>(h2_bf, w1_bf, a1_f, nullptr, MROWS, HIDP, D_MODEL, 0);
  k_gemm<<<gdh, 256, 0, stream>>>(h2_bf, w2_bf, a2_f, nullptr, MROWS, HIDP, D_MODEL, 0);
  long nmh = (long)MROWS * HIDP;
  k_silu_mul<<<(int)((nmh + 255) / 256), 256, 0, stream>>>(a1_f, a2_f, u_bf, nmh);
  k_gemm<<<gdd, 256, 0, stream>>>(u_bf, w3_bf, ffn_f, nullptr, MROWS, D_MODEL, HIDP, 0);

  long nmd = (long)MROWS * D_MODEL;
  k_add<<<(int)((nmd + 255) / 256), 256, 0, stream>>>(x2_f, ffn_f, out, nmd);
}
